// LinearAttention_26465588478053
// MI455X (gfx1250) — matrix operand order repaired and checked by reading, not compiled, not run on hardware
//
#include <hip/hip_runtime.h>
#include <hip/hip_bf16.h>

typedef __attribute__((ext_vector_type(16))) __bf16 v16bf;
typedef __attribute__((ext_vector_type(8)))  float  v8f;

#define DIMC   256
#define HIDDENC 128
#define NPOS   4096           // positions per batch (64*64)
#define BATCH  16
#define TOTPOS (BATCH * NPOS) // 65536
#define SCALE_F 0.17677669529663687f  // 32^-0.5

union Frag { uint4 u[2]; v16bf v; };

__device__ __forceinline__ unsigned short f2bf(float f) {
    unsigned u = __builtin_bit_cast(unsigned, f);
    unsigned r = (u + 0x7FFFu + ((u >> 16) & 1u)) >> 16;
    return (unsigned short)r;
}

// CDNA5 async global->LDS bulk copy (ASYNCcnt-tracked, no VGPR round trip)
__device__ __forceinline__ void async_b128(unsigned lds_addr, const void* g) {
    asm volatile("global_load_async_to_lds_b128 %0, %1, off"
                 :: "v"(lds_addr), "v"((unsigned long long)(uintptr_t)g)
                 : "memory");
}
__device__ __forceinline__ void wait_async0() {
    asm volatile("s_wait_asynccnt 0" ::: "memory");
}

// ---------------- Stage A: input RMSNorm -> bf16 [pos][c] ----------------
__global__ void k_rms_in(const float* __restrict__ x, const float* __restrict__ g,
                         unsigned short* __restrict__ xn) {
    int P = blockIdx.x * blockDim.x + threadIdx.x;   // global position
    int b = P >> 12, pl = P & 4095;
    const float* xb = x + ((size_t)b << 20) + pl;    // x[b][c][pl], stride 4096
    float ss = 0.f;
    for (int c = 0; c < DIMC; ++c) { float v = xb[(size_t)c << 12]; ss += v * v; }
    float inv = rsqrtf(ss + 1e-12f) * 16.0f;         // * sqrt(256)
    unsigned short* o = xn + ((size_t)P << 8);
    for (int c = 0; c < DIMC; ++c) o[c] = f2bf(xb[(size_t)c << 12] * inv * g[c]);
}

// ---------------- Weight conversion f32 -> bf16 (same [o][c] layout) -----
__global__ void k_wconv(const float* __restrict__ wq, const float* __restrict__ wo,
                        unsigned short* __restrict__ wqb, unsigned short* __restrict__ wob) {
    int i = blockIdx.x * blockDim.x + threadIdx.x;
    if (i < 384 * DIMC)     wqb[i] = f2bf(wq[i]);
    if (i < DIMC * HIDDENC) wob[i] = f2bf(wo[i]);
}

// ---------------- Stage B: QKV GEMM (bf16 WMMA, fp32 acc) ----------------
// block: 8 waves, M=128 (16 per wave) x N=64 shared; B tile staged in LDS
// via global_load_async_to_lds_b128. K = 256 (8 wmma steps/wave-tile).
#define BSTRIDE_Q 264   // padded LDS row stride (elements) -> conflict-free frags
__global__ void k_gemm_qkv(const unsigned short* __restrict__ xn,
                           const unsigned short* __restrict__ wq,
                           float* __restrict__ qk, unsigned short* __restrict__ vb) {
    __shared__ unsigned short bsm[64 * BSTRIDE_Q];   // 33 KB
    int tid = threadIdx.x;
    int wave = tid >> 5, lane = tid & 31;
    int mblk = blockIdx.x / 6, ntile = blockIdx.x % 6;
    int o0 = ntile * 64;
    int p0 = mblk * 128 + wave * 16;

    // stage weight tile [64 x 256] bf16 (32KB) -> padded LDS rows
    {
        const char* bsrc = (const char*)(wq + (size_t)o0 * DIMC);
        int r = tid >> 2, qd = tid & 3;                // 4 threads per row
        unsigned lbase = (unsigned)(uintptr_t)(&bsm[0]) + r * (BSTRIDE_Q * 2) + qd * 128;
        const char* gbase = bsrc + r * 512 + qd * 128;
#pragma unroll
        for (int i = 0; i < 8; ++i)
            async_b128(lbase + i * 16, gbase + i * 16);
    }
    wait_async0();
    __syncthreads();

    int m  = lane & 15;
    int kA = (lane < 16) ? 0 : 8;
    int kB = (lane < 16) ? 0 : 8;
    v8f acc[4] = {};
    const unsigned short* arow = xn + (size_t)(p0 + m) * DIMC;
    for (int kb = 0; kb < DIMC; kb += 32) {
        Frag A;
        A.u[0] = *(const uint4*)(arow + kb + kA);
        A.u[1] = *(const uint4*)(arow + kb + 16 + kA);
#pragma unroll
        for (int j = 0; j < 4; ++j) {
            const unsigned short* bcol = &bsm[(j * 16 + m) * BSTRIDE_Q + kb + kB];
            Frag B;
            B.u[0] = *(const uint4*)(bcol);
            B.u[1] = *(const uint4*)(bcol + 16);
            acc[j] = __builtin_amdgcn_wmma_f32_16x16x32_bf16(
                false, A.v, false, B.v, (short)0, acc[j], false, false);
        }
    }
#pragma unroll
    for (int j = 0; j < 4; ++j) {
        int o = o0 + j * 16 + (lane & 15);
#pragma unroll
        for (int r = 0; r < 8; ++r) {
            int p = p0 + r + ((lane >> 4) << 3);
            int b = p >> 12, pl = p & 4095;
            float val = acc[j][r];
            if (o < 256) qk[((size_t)b * 256 + o) * NPOS + pl] = val;                 // q,k fp32
            else         vb[((size_t)b * 128 + (o - 256)) * NPOS + pl] = f2bf(val);   // v bf16
        }
    }
}

// ---------------- Stage C1: q softmax over d (32) -> bf16 [b][h][n][d] ---
__global__ void k_qsoft(const float* __restrict__ qk, unsigned short* __restrict__ qb) {
    int t = blockIdx.x * blockDim.x + threadIdx.x;   // (b,h,p)
    int p = t & 4095, bh = t >> 12, h = bh & 3, b = bh >> 2;
    const float* qp = qk + ((size_t)b * 256 + h * 32) * NPOS + p;
    float vbuf[32], mx = -3.402823466e38f;
#pragma unroll
    for (int d = 0; d < 32; ++d) { float v = qp[(size_t)d << 12]; vbuf[d] = v; mx = fmaxf(mx, v); }
    float s = 0.f;
#pragma unroll
    for (int d = 0; d < 32; ++d) { float e = __expf(vbuf[d] - mx); vbuf[d] = e; s += e; }
    float inv = 1.f / s;
    unsigned short* o = qb + ((size_t)bh * NPOS + p) * 32;
#pragma unroll
    for (int d = 0; d < 32; ++d) o[d] = f2bf(vbuf[d] * inv);
}

// ---------------- Stage C2: k softmax over n (4096) -> bf16 [b][h][d][n] -
__global__ void k_ksoft(const float* __restrict__ qk, unsigned short* __restrict__ kb) {
    int row = blockIdx.x;                 // 0..2047 = b*128 + (h*32+d)
    int b = row >> 7, c = row & 127;
    const float* kr = qk + ((size_t)b * 256 + 128 + c) * NPOS;
    __shared__ float red[256];
    float mx = -3.402823466e38f;
    for (int i = threadIdx.x; i < NPOS; i += 256) mx = fmaxf(mx, kr[i]);
    red[threadIdx.x] = mx; __syncthreads();
    for (int s = 128; s > 0; s >>= 1) {
        if (threadIdx.x < (unsigned)s) red[threadIdx.x] = fmaxf(red[threadIdx.x], red[threadIdx.x + s]);
        __syncthreads();
    }
    mx = red[0]; __syncthreads();
    float sum = 0.f;
    for (int i = threadIdx.x; i < NPOS; i += 256) sum += __expf(kr[i] - mx);
    red[threadIdx.x] = sum; __syncthreads();
    for (int s = 128; s > 0; s >>= 1) {
        if (threadIdx.x < (unsigned)s) red[threadIdx.x] += red[threadIdx.x + s];
        __syncthreads();
    }
    float inv = 1.f / red[0];
    unsigned short* o = kb + (size_t)row * NPOS;
    for (int i = threadIdx.x; i < NPOS; i += 256) o[i] = f2bf(__expf(kr[i] - mx) * inv);
}

// ---------------- zero the fp32 context accumulator -----------------------
__global__ void k_zero(float* __restrict__ ctf) {
    ctf[blockIdx.x * blockDim.x + threadIdx.x] = 0.f;
}

// ---------------- Stage D: context = k' @ v^T, split-K + atomics ----------
// grid = 64 (b,h) x 16 K-chunks, one wave each; partial 32x32 via 2x2 WMMA
__global__ void k_context(const unsigned short* __restrict__ kb,
                          const unsigned short* __restrict__ vb,
                          float* __restrict__ ctf) {
    int bh = blockIdx.x >> 4, kc = blockIdx.x & 15, lane = threadIdx.x;
    const unsigned short* kbase = kb + (size_t)bh * 32 * NPOS;
    const unsigned short* vbase = vb + (size_t)bh * 32 * NPOS;
    int m  = lane & 15;
    int kA = (lane < 16) ? 0 : 8;
    int kB = (lane < 16) ? 0 : 8;
    v8f acc[2][2] = {};
    int k0 = kc * 256;
    for (int kb0 = k0; kb0 < k0 + 256; kb0 += 32) {
        Frag A[2], B[2];
#pragma unroll
        for (int i = 0; i < 2; ++i) {
            const unsigned short* ar = kbase + (size_t)(i * 16 + m) * NPOS + kb0;
            A[i].u[0] = *(const uint4*)(ar + kA);
            A[i].u[1] = *(const uint4*)(ar + 16 + kA);
            const unsigned short* br = vbase + (size_t)(i * 16 + m) * NPOS + kb0 + kB;
            B[i].u[0] = *(const uint4*)(br);
            B[i].u[1] = *(const uint4*)(br + 16);
        }
#pragma unroll
        for (int i = 0; i < 2; ++i)
#pragma unroll
            for (int j = 0; j < 2; ++j)
                acc[i][j] = __builtin_amdgcn_wmma_f32_16x16x32_bf16(
                    false, A[i].v, false, B[j].v, (short)0, acc[i][j], false, false);
    }
#pragma unroll
    for (int i = 0; i < 2; ++i)
#pragma unroll
        for (int j = 0; j < 2; ++j)
#pragma unroll
            for (int r = 0; r < 8; ++r) {
                int d = i * 16 + r + ((lane >> 4) << 3);
                int e = j * 16 + (lane & 15);
                atomicAdd(&ctf[((size_t)bh * 32 + d) * 32 + e], acc[i][j][r]);
            }
}

// ---------------- convert context fp32 [d][e] -> bf16 [e][d], *SCALE ------
__global__ void k_ctconv(const float* __restrict__ ctf, unsigned short* __restrict__ ct) {
    int i = blockIdx.x * blockDim.x + threadIdx.x;   // output index
    int d = i & 31, e = (i >> 5) & 31, bh = i >> 10;
    ct[i] = f2bf(ctf[(size_t)bh * 1024 + d * 32 + e] * SCALE_F);
}

// ---------------- Stage E: out = q' @ ct (per b,h, 16-pos tiles) ----------
__global__ void k_attnout(const unsigned short* __restrict__ qb,
                          const unsigned short* __restrict__ ct,
                          unsigned short* __restrict__ oa) {
    int wid  = blockIdx.x * (blockDim.x >> 5) + (threadIdx.x >> 5);
    int lane = threadIdx.x & 31;
    int pt = wid & 255, bh = wid >> 8, h = bh & 3, b = bh >> 2;
    int p0 = pt * 16;
    int m  = lane & 15;
    int kA = (lane < 16) ? 0 : 8;
    int kB = (lane < 16) ? 0 : 8;
    Frag A, B[2];
    const unsigned short* ar = qb + ((size_t)bh * NPOS + p0 + m) * 32;
    A.u[0] = *(const uint4*)(ar + kA);
    A.u[1] = *(const uint4*)(ar + 16 + kA);
#pragma unroll
    for (int j = 0; j < 2; ++j) {
        const unsigned short* br = ct + ((size_t)bh * 32 + j * 16 + m) * 32 + kB;
        B[j].u[0] = *(const uint4*)(br);
        B[j].u[1] = *(const uint4*)(br + 16);
    }
    v8f acc[2] = {};
#pragma unroll
    for (int j = 0; j < 2; ++j)
        acc[j] = __builtin_amdgcn_wmma_f32_16x16x32_bf16(
            false, A.v, false, B[j].v, (short)0, acc[j], false, false);
#pragma unroll
    for (int j = 0; j < 2; ++j)
#pragma unroll
        for (int r = 0; r < 8; ++r) {
            int p = p0 + r + ((lane >> 4) << 3);
            int e = h * 32 + j * 16 + (lane & 15);
            oa[((size_t)b * NPOS + p) * HIDDENC + e] = f2bf(acc[j][r]);
        }
}

// ---------------- Stage F: out projection GEMM + bias ---------------------
// block: 8 waves share N=64, M=128; B tile (64x128 bf16) staged via async LDS
#define BSTRIDE_O 136   // padded LDS row stride (elements)
__global__ void k_gemm_out(const unsigned short* __restrict__ oa,
                           const unsigned short* __restrict__ wo,
                           const float* __restrict__ bias, float* __restrict__ o2) {
    __shared__ unsigned short bsm[64 * BSTRIDE_O];   // 17 KB
    int tid = threadIdx.x;
    int wave = tid >> 5, lane = tid & 31;
    int ntile = blockIdx.x & 3, mblk = blockIdx.x >> 2;
    int o0 = ntile * 64;
    int p0 = mblk * 128 + wave * 16;

    {
        const char* bsrc = (const char*)(wo + (size_t)o0 * HIDDENC);
        int r = tid >> 2, qd = tid & 3;                // 4 threads per row, 64B each
        unsigned lbase = (unsigned)(uintptr_t)(&bsm[0]) + r * (BSTRIDE_O * 2) + qd * 64;
        const char* gbase = bsrc + r * 256 + qd * 64;
#pragma unroll
        for (int i = 0; i < 4; ++i)
            async_b128(lbase + i * 16, gbase + i * 16);
    }
    wait_async0();
    __syncthreads();

    int m  = lane & 15;
    int kA = (lane < 16) ? 0 : 8;
    int kB = (lane < 16) ? 0 : 8;
    v8f acc[4] = {};
    const unsigned short* arow = oa + (size_t)(p0 + m) * HIDDENC;
    for (int kb = 0; kb < HIDDENC; kb += 32) {
        Frag A;
        A.u[0] = *(const uint4*)(arow + kb + kA);
        A.u[1] = *(const uint4*)(arow + kb + 16 + kA);
#pragma unroll
        for (int j = 0; j < 4; ++j) {
            const unsigned short* bcol = &bsm[(j * 16 + m) * BSTRIDE_O + kb + kB];
            Frag B;
            B.u[0] = *(const uint4*)(bcol);
            B.u[1] = *(const uint4*)(bcol + 16);
            acc[j] = __builtin_amdgcn_wmma_f32_16x16x32_bf16(
                false, A.v, false, B.v, (short)0, acc[j], false, false);
        }
    }
#pragma unroll
    for (int j = 0; j < 4; ++j) {
        int o = o0 + j * 16 + (lane & 15);
        float bo = bias[o];
#pragma unroll
        for (int r = 0; r < 8; ++r) {
            int p = p0 + r + ((lane >> 4) << 3);
            o2[(size_t)p * DIMC + o] = acc[j][r] + bo;
        }
    }
}

// ---------------- Stage G: output RMSNorm -> NCHW layout ------------------
__global__ void k_rms_out(const float* __restrict__ o2, const float* __restrict__ g,
                          float* __restrict__ out) {
    int P = blockIdx.x * blockDim.x + threadIdx.x;
    int b = P >> 12, pl = P & 4095;
    const float* r = o2 + (size_t)P * DIMC;
    float ss = 0.f;
    for (int c = 0; c < DIMC; ++c) { float v = r[c]; ss += v * v; }
    float inv = rsqrtf(ss + 1e-12f) * 16.0f;
    for (int c = 0; c < DIMC; ++c)
        out[((size_t)b * DIMC + c) * NPOS + pl] = r[c] * inv * g[c];
}

extern "C" void kernel_launch(void* const* d_in, const int* in_sizes, int n_in,
                              void* d_out, int out_size, void* d_ws, size_t ws_size,
                              hipStream_t stream) {
    (void)in_sizes; (void)n_in; (void)out_size; (void)ws_size;
    const float* x     = (const float*)d_in[0];
    const float* Wqkv  = (const float*)d_in[1];
    const float* Wout  = (const float*)d_in[2];
    const float* bout  = (const float*)d_in[3];
    const float* gnorm = (const float*)d_in[4];
    const float* gout  = (const float*)d_in[5];
    float* out = (float*)d_out;

    char* p = (char*)d_ws;
    // region 0: q,k fp32 (64MB) -- later reused as out-proj fp32 result
    float* qk = (float*)p;
    float* o2 = (float*)p;
    // region 1: xn bf16 (32MB) -- later reused: q' (16MB) + k' (16MB)
    unsigned short* xn   = (unsigned short*)(p + 67108864);
    unsigned short* qb   = xn;
    unsigned short* kbuf = (unsigned short*)(p + 67108864 + 16777216);
    // region 2: v bf16 (16MB)
    unsigned short* vb   = (unsigned short*)(p + 100663296);
    // region 3: attention output bf16 (16MB)
    unsigned short* oa   = (unsigned short*)(p + 117440512);
    // region 4: context bf16 (128KB) + bf16 weights + context fp32 (256KB)
    unsigned short* ct   = (unsigned short*)(p + 134217728);
    unsigned short* wqb  = ct + 64 * 32 * 32;
    unsigned short* wob  = wqb + 384 * 256;
    float*          ctf  = (float*)(wob + 256 * 128);

    k_rms_in  <<<TOTPOS / 256, 256, 0, stream>>>(x, gnorm, xn);
    k_wconv   <<<384, 256, 0, stream>>>(Wqkv, Wout, wqb, wob);
    k_gemm_qkv<<<3072, 256, 0, stream>>>(xn, wqb, qk, vb);          // 512 Mblk x 6 Ntile
    k_qsoft   <<<1024, 256, 0, stream>>>(qk, qb);
    k_ksoft   <<<2048, 256, 0, stream>>>(qk, kbuf);
    k_zero    <<<256, 256, 0, stream>>>(ctf);
    k_context <<<1024, 32, 0, stream>>>(kbuf, vb, ctf);             // 64 bh x 16 K-chunks
    k_ctconv  <<<256, 256, 0, stream>>>(ctf, ct);
    k_attnout <<<2048, 256, 0, stream>>>(qb, ct, oa);
    k_gemm_out<<<2048, 256, 0, stream>>>(oa, wob, bout, o2);        // 512 Mblk x 4 Ntile
    k_rms_out <<<TOTPOS / 256, 256, 0, stream>>>(o2, gout, out);
}
